// SpatialCrossAttention_15015205667216
// MI455X (gfx1250) — compile-verified, matrix operand order checked by reading, not run on hardware
//
#include <hip/hip_runtime.h>
#include <hip/hip_bf16.h>

// SpatialCrossAttention (BEVFormer-style) for gfx1250 / MI455X.
// Dense projections: LDS-staged bf16 WMMA (v_wmma_f32_16x16x32_bf16, wave32).
//   - 256/192-thread block = one WGP tile strip: 16 rows x full output width
//   - weights staged ONCE per block into LDS, pre-swizzled into WMMA fragment
//     order (contiguous 32B per lane fragment -> ds_load_b128 pairs)
//   - A strip (16 x 128) staged once and shared by all waves in the block
// Bilinear sampling: wave-per-(query,head) gather kernel (lane = channel).

#define NUM_HEADS  4
#define NUM_POINTS 8
#define CDIM       128
#define DHDIM      32            // CDIM / NUM_HEADS
#define QOUTC      96            // 64 offset cols (Wo) + 32 attn cols (Wa)
#define KCHUNKS    4             // 128 / 32

typedef __attribute__((ext_vector_type(16))) __bf16 v16bf;
typedef __attribute__((ext_vector_type(8)))  float  v8f;

static __device__ __forceinline__ __bf16 tobf(float f) { return (__bf16)f; }

// fragment element -> K offset within a 32-wide chunk
static __device__ __forceinline__ int kofs(int i, int kbase) {
  return kbase + (i & 7) + ((i & 8) ? 16 : 0);
}

// ------------------------------------------------------------------
// GEMM 1: value projection.  grid = (M/16, S), block = 256 (8 waves).
// A = value[s] [M,128] f32, B = Wv [128,128], bias bv.
// Output layout: value_hw[S][H][M][Dh].
// ------------------------------------------------------------------
__global__ void k_value_proj(const float* __restrict__ val,
                             const float* __restrict__ Wv,
                             const float* __restrict__ bv,
                             float* __restrict__ value_hw,
                             int M) {
  __shared__ __align__(32) __bf16 sB[8 * KCHUNKS * 32 * 16];   // 32 KB
  __shared__ __align__(32) __bf16 sA[KCHUNKS * 32 * 16];       // 4 KB
  const int tid = threadIdx.x;
  const int s   = blockIdx.y;
  const int m0  = blockIdx.x * 16;
  const float* A = val + (size_t)s * M * CDIM;

  // stage full Wv in WMMA fragment order (bf16 pairs, consecutive K)
  for (int e = tid; e < 8 * KCHUNKS * 32 * 8; e += blockDim.x) {
    const int i2 = (e & 7) * 2;
    const int l  = (e >> 3) & 31;
    const int c  = (e >> 8) & 3;
    const int t  = e >> 10;
    const int k  = c * 32 + kofs(i2, (l >> 4) * 8);
    const int n  = t * 16 + (l & 15);
    const int o  = (((t * KCHUNKS + c) * 32) + l) * 16 + i2;
    sB[o]     = tobf(Wv[(size_t)k * CDIM + n]);
    sB[o + 1] = tobf(Wv[(size_t)(k + 1) * CDIM + n]);
  }
  // stage the 16-row A strip (shared by all 8 waves)
  for (int e = tid; e < KCHUNKS * 32 * 8; e += blockDim.x) {
    const int i2  = (e & 7) * 2;
    const int l   = (e >> 3) & 31;
    const int c   = e >> 8;
    const int k   = c * 32 + kofs(i2, (l >> 4) * 8);
    const int row = m0 + (l & 15);
    const float2 f2 = *(const float2*)&A[(size_t)row * CDIM + k];
    const int o = ((c * 32) + l) * 16 + i2;
    sA[o]     = tobf(f2.x);
    sA[o + 1] = tobf(f2.y);
  }
  __syncthreads();

  const int wv   = tid >> 5;        // n-tile index (0..7)
  const int lane = tid & 31;
  v8f acc = {};
#pragma unroll
  for (int c = 0; c < KCHUNKS; ++c) {
    const v16bf a = *(const v16bf*)&sA[(c * 32 + lane) * 16];
    const v16bf b = *(const v16bf*)&sB[((wv * KCHUNKS + c) * 32 + lane) * 16];
    acc = __builtin_amdgcn_wmma_f32_16x16x32_bf16(false, a, false, b,
                                                  (short)0, acc, false, false);
  }
  const int hl = lane >> 4, l15 = lane & 15;
  const int ncol = wv * 16 + l15;
  const float bias = bv[ncol];
  const int h = ncol >> 5, dh = ncol & 31;
  float* obase = value_hw + (size_t)(s * NUM_HEADS + h) * M * DHDIM;
#pragma unroll
  for (int r = 0; r < 8; ++r) {
    const int mm = m0 + r + hl * 8;
    obase[(size_t)mm * DHDIM + dh] = acc[r] + bias;
  }
}

// ------------------------------------------------------------------
// GEMM 2: query projections.  grid = (Nq/16), block = 192 (6 waves).
// A = (query + query_pos) fused during staging.
// B = concat(Wo [128,64], Wa [128,32]) -> 96 cols (tile-aligned split).
// Output: qproj [Nq, 96].
// ------------------------------------------------------------------
__global__ void k_qproj(const float* __restrict__ query,
                        const float* __restrict__ query_pos,
                        const float* __restrict__ Wo, const float* __restrict__ bo,
                        const float* __restrict__ Wa, const float* __restrict__ ba,
                        float* __restrict__ qproj, int Nq) {
  __shared__ __align__(32) __bf16 sB[6 * KCHUNKS * 32 * 16];   // 24 KB
  __shared__ __align__(32) __bf16 sA[KCHUNKS * 32 * 16];       // 4 KB
  const int tid = threadIdx.x;
  const int m0  = blockIdx.x * 16;

  for (int e = tid; e < 6 * KCHUNKS * 32 * 8; e += blockDim.x) {
    const int i2 = (e & 7) * 2;
    const int l  = (e >> 3) & 31;
    const int c  = (e >> 8) & 3;
    const int t  = e >> 10;
    const int k  = c * 32 + kofs(i2, (l >> 4) * 8);
    const int n  = t * 16 + (l & 15);
    const float f0 = (n < 64) ? Wo[(size_t)k * 64 + n]       : Wa[(size_t)k * 32 + (n - 64)];
    const float f1 = (n < 64) ? Wo[(size_t)(k + 1) * 64 + n] : Wa[(size_t)(k + 1) * 32 + (n - 64)];
    const int o = (((t * KCHUNKS + c) * 32) + l) * 16 + i2;
    sB[o]     = tobf(f0);
    sB[o + 1] = tobf(f1);
  }
  for (int e = tid; e < KCHUNKS * 32 * 8; e += blockDim.x) {
    const int i2  = (e & 7) * 2;
    const int l   = (e >> 3) & 31;
    const int c   = e >> 8;
    const int k   = c * 32 + kofs(i2, (l >> 4) * 8);
    const int row = m0 + (l & 15);
    const float2 q2 = *(const float2*)&query[(size_t)row * CDIM + k];
    const float2 p2 = *(const float2*)&query_pos[(size_t)row * CDIM + k];
    const int o = ((c * 32) + l) * 16 + i2;
    sA[o]     = tobf(q2.x + p2.x);
    sA[o + 1] = tobf(q2.y + p2.y);
  }
  __syncthreads();

  const int wv   = tid >> 5;        // n-tile (0..5)
  const int lane = tid & 31;
  v8f acc = {};
#pragma unroll
  for (int c = 0; c < KCHUNKS; ++c) {
    const v16bf a = *(const v16bf*)&sA[(c * 32 + lane) * 16];
    const v16bf b = *(const v16bf*)&sB[((wv * KCHUNKS + c) * 32 + lane) * 16];
    acc = __builtin_amdgcn_wmma_f32_16x16x32_bf16(false, a, false, b,
                                                  (short)0, acc, false, false);
  }
  const int hl = lane >> 4, l15 = lane & 15;
  const int ncol = wv * 16 + l15;
  const float bias = (ncol < 64) ? bo[ncol] : ba[ncol - 64];
#pragma unroll
  for (int r = 0; r < 8; ++r) {
    const int gm = m0 + r + hl * 8;
    qproj[(size_t)gm * QOUTC + ncol] = acc[r] + bias;
  }
}

// ------------------------------------------------------------------
// Softmax over the 8 points per (query, head).
// ------------------------------------------------------------------
__global__ void k_softmax(const float* __restrict__ qproj,
                          float* __restrict__ attw, int Nq) {
  const int t = blockIdx.x * blockDim.x + threadIdx.x;
  if (t >= Nq * NUM_HEADS) return;
  const int n = t / NUM_HEADS, h = t % NUM_HEADS;
  const float* src = qproj + (size_t)n * QOUTC + 64 + h * NUM_POINTS;
  float mx = src[0];
#pragma unroll
  for (int p = 1; p < NUM_POINTS; ++p) mx = fmaxf(mx, src[p]);
  float e[NUM_POINTS], sum = 0.f;
#pragma unroll
  for (int p = 0; p < NUM_POINTS; ++p) { e[p] = __expf(src[p] - mx); sum += e[p]; }
  const float inv = 1.f / sum;
  float* dst = attw + ((size_t)n * NUM_HEADS + h) * NUM_POINTS;
#pragma unroll
  for (int p = 0; p < NUM_POINTS; ++p) dst[p] = e[p] * inv;
}

// ------------------------------------------------------------------
// Bilinear sampling + point weighting + camera mask/normalize.
// One wave per (query, head); lane = Dh channel (coalesced 32-lane taps).
// ------------------------------------------------------------------
__global__ void k_sample(const float* __restrict__ value_hw,
                         const float* __restrict__ qproj,
                         const float* __restrict__ attw,
                         const float* __restrict__ refpts,      // [S,1,Nq,4,2]
                         const unsigned char* __restrict__ bev, // [S,1,Nq,4] bool
                         const long long* __restrict__ spatial, // [1,2] int64
                         float* __restrict__ slots,
                         int Nq, int S, int M) {
  const int wave = (blockIdx.x * blockDim.x + threadIdx.x) >> 5;
  const int lane = threadIdx.x & 31;
  if (wave >= Nq * NUM_HEADS) return;
  const int n = wave >> 2;
  const int h = wave & 3;
  const int Hf = (int)spatial[0];
  const int Wf = (int)spatial[1];

  float cnt = 0.f;
  unsigned vis = 0u;
  for (int s = 0; s < S; ++s) {
    const unsigned char* mb = bev + ((size_t)s * Nq + n) * 4;
    if (mb[0] | mb[1] | mb[2] | mb[3]) { vis |= (1u << s); cnt += 1.f; }
  }
  const float invc = 1.f / fmaxf(cnt, 1.f);

  const float* offrow = qproj + (size_t)n * QOUTC;
  const float* wrow   = attw + ((size_t)n * NUM_HEADS + h) * NUM_POINTS;

  float acc = 0.f;
  for (int s = 0; s < S; ++s) {
    if (!(vis & (1u << s))) continue;
    const float* vbase = value_hw + (size_t)(s * NUM_HEADS + h) * M * DHDIM;
    const float* rp    = refpts + ((size_t)s * Nq + n) * 8;
#pragma unroll
    for (int p = 0; p < NUM_POINTS; ++p) {
      const int d = p & 3;
      const float x = rp[d * 2 + 0] * (float)Wf + offrow[(h * NUM_POINTS + p) * 2 + 0] - 0.5f;
      const float y = rp[d * 2 + 1] * (float)Hf + offrow[(h * NUM_POINTS + p) * 2 + 1] - 0.5f;
      const float x0f = floorf(x), y0f = floorf(y);
      const float lx = x - x0f, ly = y - y0f;
      const int x0 = (int)x0f, y0 = (int)y0f;
      const bool xin0 = (x0 >= 0) && (x0 < Wf);
      const bool xin1 = (x0 + 1 >= 0) && (x0 + 1 < Wf);
      float v00 = 0.f, v10 = 0.f, v01 = 0.f, v11 = 0.f;
      if (y0 >= 0 && y0 < Hf) {
        const float* row = vbase + (size_t)y0 * Wf * DHDIM;
        if (xin0) v00 = row[(size_t)x0 * DHDIM + lane];
        if (xin1) v10 = row[(size_t)(x0 + 1) * DHDIM + lane];
      }
      if (y0 + 1 >= 0 && y0 + 1 < Hf) {
        const float* row = vbase + (size_t)(y0 + 1) * Wf * DHDIM;
        if (xin0) v01 = row[(size_t)x0 * DHDIM + lane];
        if (xin1) v11 = row[(size_t)(x0 + 1) * DHDIM + lane];
      }
      const float sample = v00 * (1.f - lx) * (1.f - ly) + v10 * lx * (1.f - ly)
                         + v01 * (1.f - lx) * ly          + v11 * lx * ly;
      acc = fmaf(wrow[p], sample, acc);
    }
  }
  slots[(size_t)n * CDIM + h * DHDIM + lane] = acc * invc;
}

// ------------------------------------------------------------------
// GEMM 3: out = slots @ Wp + bp + query.  grid = (Nq/16), block = 256.
// ------------------------------------------------------------------
__global__ void k_outproj(const float* __restrict__ slots,
                          const float* __restrict__ Wp, const float* __restrict__ bp,
                          const float* __restrict__ query,
                          float* __restrict__ out, int Nq) {
  __shared__ __align__(32) __bf16 sB[8 * KCHUNKS * 32 * 16];   // 32 KB
  __shared__ __align__(32) __bf16 sA[KCHUNKS * 32 * 16];       // 4 KB
  const int tid = threadIdx.x;
  const int m0  = blockIdx.x * 16;

  for (int e = tid; e < 8 * KCHUNKS * 32 * 8; e += blockDim.x) {
    const int i2 = (e & 7) * 2;
    const int l  = (e >> 3) & 31;
    const int c  = (e >> 8) & 3;
    const int t  = e >> 10;
    const int k  = c * 32 + kofs(i2, (l >> 4) * 8);
    const int n  = t * 16 + (l & 15);
    const int o  = (((t * KCHUNKS + c) * 32) + l) * 16 + i2;
    sB[o]     = tobf(Wp[(size_t)k * CDIM + n]);
    sB[o + 1] = tobf(Wp[(size_t)(k + 1) * CDIM + n]);
  }
  for (int e = tid; e < KCHUNKS * 32 * 8; e += blockDim.x) {
    const int i2  = (e & 7) * 2;
    const int l   = (e >> 3) & 31;
    const int c   = e >> 8;
    const int k   = c * 32 + kofs(i2, (l >> 4) * 8);
    const int row = m0 + (l & 15);
    const float2 f2 = *(const float2*)&slots[(size_t)row * CDIM + k];
    const int o = ((c * 32) + l) * 16 + i2;
    sA[o]     = tobf(f2.x);
    sA[o + 1] = tobf(f2.y);
  }
  __syncthreads();

  const int wv   = tid >> 5;
  const int lane = tid & 31;
  v8f acc = {};
#pragma unroll
  for (int c = 0; c < KCHUNKS; ++c) {
    const v16bf a = *(const v16bf*)&sA[(c * 32 + lane) * 16];
    const v16bf b = *(const v16bf*)&sB[((wv * KCHUNKS + c) * 32 + lane) * 16];
    acc = __builtin_amdgcn_wmma_f32_16x16x32_bf16(false, a, false, b,
                                                  (short)0, acc, false, false);
  }
  const int hl = lane >> 4, l15 = lane & 15;
  const int ncol = wv * 16 + l15;
  const float bias = bp[ncol];
#pragma unroll
  for (int r = 0; r < 8; ++r) {
    const int gm = m0 + r + hl * 8;
    out[(size_t)gm * CDIM + ncol] = acc[r] + bias + query[(size_t)gm * CDIM + ncol];
  }
}

// ------------------------------------------------------------------
extern "C" void kernel_launch(void* const* d_in, const int* in_sizes, int n_in,
                              void* d_out, int out_size, void* d_ws, size_t ws_size,
                              hipStream_t stream) {
  const float* query     = (const float*)d_in[0];
  // d_in[1] = key : unused by the reference math
  const float* value     = (const float*)d_in[2];
  const float* query_pos = (const float*)d_in[3];
  const float* refpts    = (const float*)d_in[4];
  const float* Wv = (const float*)d_in[5];
  const float* bv = (const float*)d_in[6];
  const float* Wo = (const float*)d_in[7];
  const float* bo = (const float*)d_in[8];
  const float* Wa = (const float*)d_in[9];
  const float* ba = (const float*)d_in[10];
  const float* Wp = (const float*)d_in[11];
  const float* bp = (const float*)d_in[12];
  const long long*     spatial = (const long long*)d_in[13];     // int64 [Hf,Wf]
  const unsigned char* bev     = (const unsigned char*)d_in[14]; // bool bytes

  const int C  = CDIM;
  const int Nq = in_sizes[0] / C;          // 10000
  const int SM = in_sizes[2] / C;          // S*M = 22080
  const int S  = in_sizes[4] / (Nq * 8);   // refpts = S*Nq*D*2, D=4 -> S=6
  const int M  = SM / S;                   // 3680

  // workspace layout (floats)
  float* ws       = (float*)d_ws;
  float* value_hw = ws;                                                // S*4*M*32
  float* qproj    = value_hw + (size_t)S * NUM_HEADS * M * DHDIM;      // Nq*96
  float* attw     = qproj + (size_t)Nq * QOUTC;                        // Nq*32
  float* slots    = attw + (size_t)Nq * NUM_HEADS * NUM_POINTS;        // Nq*128

  // 1) value projection (WMMA, LDS-staged); camera index in grid.y (no div)
  k_value_proj<<<dim3(M / 16, S), 256, 0, stream>>>(value, Wv, bv, value_hw, M);

  // 2) fused query projections (WMMA, LDS-staged)
  k_qproj<<<dim3(Nq / 16), 192, 0, stream>>>(query, query_pos, Wo, bo, Wa, ba,
                                             qproj, Nq);

  // 3) per-(query,head) softmax over points
  {
    const int t = Nq * NUM_HEADS;
    k_softmax<<<(t + 255) / 256, 256, 0, stream>>>(qproj, attw, Nq);
  }

  // 4) bilinear sampling + mask accumulate (wave per (query,head))
  {
    const int waves  = Nq * NUM_HEADS;
    const int blocks = (waves + 7) / 8;
    k_sample<<<blocks, 256, 0, stream>>>(value_hw, qproj, attw, refpts, bev,
                                         spatial, slots, Nq, S, M);
  }

  // 5) output projection + bias + residual (WMMA, LDS-staged)
  k_outproj<<<dim3(Nq / 16), 256, 0, stream>>>(slots, Wp, bp, query,
                                               (float*)d_out, Nq);
}
